// MemBlock_87316685127952
// MI455X (gfx1250) — compile-verified, matrix operand order checked by reading, not run on hardware
//
#include <hip/hip_runtime.h>

typedef __bf16 bf16;
typedef __attribute__((ext_vector_type(8)))  bf16  bf16x8;   // 16 bytes
typedef __attribute__((ext_vector_type(16))) bf16  v16bf;    // WMMA A/B frag
typedef __attribute__((ext_vector_type(8)))  float v8f;      // WMMA C/D frag
typedef __attribute__((ext_vector_type(4)))  unsigned int u32x4;
typedef __attribute__((ext_vector_type(8)))  int i32x8;
typedef __attribute__((ext_vector_type(4)))  int i32x4;

constexpr int Bc = 4, Tc = 1024, Hc = 1024, Kh = 16, HD = 64;
constexpr int Mrows = Bc * Tc;       // 4096
constexpr int H4 = 4 * Hc;           // 4096

// Assemble a 16x32 bf16 A/B WMMA fragment from two aligned 16-byte chunks.
// ISA layout (16-bit A 16x32): element i -> k = (i<8?0:16) + half*8 + (i&7).
static __device__ inline v16bf load_frag(const bf16* p_lo, const bf16* p_hi) {
  bf16x8 lo = *(const bf16x8*)p_lo;
  bf16x8 hi = *(const bf16x8*)p_hi;
  v16bf r;
#pragma unroll
  for (int i = 0; i < 8; ++i) { r[i] = lo[i]; r[i + 8] = hi[i]; }
  return r;
}

static __device__ inline v8f wmma_bf16(v16bf a, v16bf b, v8f c) {
  return __builtin_amdgcn_wmma_f32_16x16x32_bf16(false, a, false, b, (short)0, c,
                                                 false, false);
}

// TDM: load a 2D tile (32 k-elements x nrows rows, bf16) from global into LDS,
// padding each 64B row with 16B so LDS row pitch = 80B (40 bf16) -> conflict-free
// 16B fragment reads.  Descriptor per CDNA5 ISA ch. 8 (D# groups 0/1); this
// toolchain's builtin takes 6 args (g0, g1, g2, g3, extra group, cpol).
static __device__ inline void tdm_load_tile(unsigned lds_addr, const bf16* gptr,
                                            int Kd, int nrows, int tensor_rows) {
  const unsigned long long va = (unsigned long long)(size_t)gptr;
  u32x4 g0;
  g0[0] = 1u;                                         // count=1 (valid), no gather
  g0[1] = lds_addr;                                   // LDS byte address
  g0[2] = (unsigned)va;                               // global_addr[31:0]
  g0[3] = ((unsigned)(va >> 32) & 0x01FFFFFFu) | (2u << 30);  // addr[56:32]|type=2
  i32x8 g1;
  g1[0] = (1 << 16)      // data_size = 2 bytes
        | (1 << 20)      // pad_enable
        | (3 << 22)      // pad_interval: every 16 DWORDs (= 32 bf16 row)
        | (3 << 25);     // pad_amount: 4 DWORDs (16B)
  g1[1] = (int)(((unsigned)Kd & 0xFFFFu) << 16);                       // dim0 lo
  g1[2] = (int)(((unsigned)Kd >> 16) | (((unsigned)tensor_rows & 0xFFFFu) << 16));
  g1[3] = (int)((((unsigned)tensor_rows >> 16) & 0xFFFFu) | (32u << 16)); // tile_dim0=32
  g1[4] = nrows;                                      // tile_dim1, tile_dim2=0
  g1[5] = Kd;                                         // tensor_dim0_stride lo
  g1[6] = 0;
  g1[7] = 0;
  i32x4 z4 = {0, 0, 0, 0};
  i32x8 z8 = {0, 0, 0, 0, 0, 0, 0, 0};
  __builtin_amdgcn_tensor_load_to_lds(g0, g1, z4, z4, z8, 0);
}

// ---------------- elementwise kernels ----------------
__global__ void cvt_t_kernel(const float* __restrict__ in,  // Kd x N, row-major
                             bf16* __restrict__ out,        // N x Kd (transposed)
                             int Kd, int N) {
  int i = blockIdx.x * blockDim.x + threadIdx.x;
  if (i < Kd * N) {
    int k = i / N, n = i - k * N;
    out[(size_t)n * Kd + k] = (bf16)in[i];
  }
}

__global__ void add_kernel(const float* __restrict__ a, const float* __restrict__ b,
                           float* __restrict__ out, int n) {
  int i = blockIdx.x * blockDim.x + threadIdx.x;
  if (i < n) out[i] = a[i] + b[i];
}

// ---------------- LayerNorm (fp32 in, bf16 out) ----------------
__global__ __launch_bounds__(256) void ln_kernel(const float* __restrict__ x,
                                                 const float* __restrict__ g,
                                                 const float* __restrict__ b,
                                                 bf16* __restrict__ out) {
  __shared__ float red[16];
  const int row = blockIdx.x;
  const int tid = threadIdx.x;
  const float* xr = x + (size_t)row * Hc;
  float s = 0.f, ss = 0.f;
#pragma unroll
  for (int i = 0; i < 4; ++i) {
    float v = xr[tid + i * 256];
    s += v; ss += v * v;
  }
#pragma unroll
  for (int m = 16; m >= 1; m >>= 1) {
    s  += __shfl_xor(s,  m, 32);
    ss += __shfl_xor(ss, m, 32);
  }
  if ((tid & 31) == 0) { red[tid >> 5] = s; red[8 + (tid >> 5)] = ss; }
  __syncthreads();
  float ts = 0.f, tss = 0.f;
#pragma unroll
  for (int w = 0; w < 8; ++w) { ts += red[w]; tss += red[8 + w]; }
  const float mu = ts * (1.0f / Hc);
  const float var = tss * (1.0f / Hc) - mu * mu;
  const float rs = rsqrtf(var + 1e-5f);
#pragma unroll
  for (int i = 0; i < 4; ++i) {
    int col = tid + i * 256;
    float v = (xr[col] - mu) * rs * g[col] + b[col];
    out[(size_t)row * Hc + col] = (bf16)v;
  }
}

// ---------------- tiled bf16 GEMM: C = act(A @ Wt^T + bias) [+ resid] ----------
// A:  M x Kd  (bf16, row-major)
// Wt: N x Kd  (bf16, row-major = original W transposed)
// Block tile 128x128, 8 waves (4 row-stripes x 2 col-stripes), per-wave 2x4
// 16x16 frags, BK=32, double-buffered LDS staged by the Tensor Data Mover.
__global__ __launch_bounds__(256) void gemm_bf16_kernel(
    const bf16* __restrict__ A, const bf16* __restrict__ Wt,
    const float* __restrict__ bias, const float* __restrict__ resid,
    bf16* __restrict__ outb, float* __restrict__ outf,
    int M, int N, int Kd, int act) {
  __shared__ __align__(16) bf16 As[2][128 * 40];  // [m][k] pitch 40 (80B)
  __shared__ __align__(16) bf16 Bs[2][128 * 40];  // [n][k] pitch 40

  const int tid  = threadIdx.x;
  const int lane = tid & 31;
  const int wid  = tid >> 5;
  const int half = lane >> 4;
  const int nn   = lane & 15;
  const int wm   = wid & 3;   // 0..3 -> 32-row stripes
  const int wn   = wid >> 2;  // 0..1 -> 64-col stripes
  const int rowb = blockIdx.y * 128;
  const int colb = blockIdx.x * 128;

  const bf16* Ab = A  + (size_t)rowb * Kd;
  const bf16* Bb = Wt + (size_t)colb * Kd;

  v8f acc[2][4];
#pragma unroll
  for (int ti = 0; ti < 2; ++ti)
#pragma unroll
    for (int tj = 0; tj < 4; ++tj)
#pragma unroll
      for (int r = 0; r < 8; ++r) acc[ti][tj][r] = 0.f;

  const int nsteps = Kd >> 5;  // Kd / 32
  if (wid == 0) {  // prologue: stage 0
    tdm_load_tile((unsigned)(size_t)&As[0][0], Ab, Kd, 128, M);
    tdm_load_tile((unsigned)(size_t)&Bs[0][0], Bb, Kd, 128, N);
  }

  for (int s = 0; s < nsteps; ++s) {
    const int cur = s & 1;
    if (wid == 0) {
      if (s + 1 < nsteps) {  // stage s+1 into the other buffer
        const int nxt = cur ^ 1;
        tdm_load_tile((unsigned)(size_t)&As[nxt][0], Ab + (s + 1) * 32, Kd, 128, M);
        tdm_load_tile((unsigned)(size_t)&Bs[nxt][0], Bb + (s + 1) * 32, Kd, 128, N);
        __builtin_amdgcn_s_wait_tensorcnt((short)2);  // stage s complete
      } else {
        __builtin_amdgcn_s_wait_tensorcnt((short)0);
      }
    }
    __syncthreads();  // stage s visible to all waves

    v16bf af[2], bfm[4];
#pragma unroll
    for (int ti = 0; ti < 2; ++ti) {
      const bf16* r = &As[cur][(wm * 32 + ti * 16 + nn) * 40];
      af[ti] = load_frag(r + half * 8, r + 16 + half * 8);
    }
#pragma unroll
    for (int tj = 0; tj < 4; ++tj) {
      const bf16* r = &Bs[cur][(wn * 64 + tj * 16 + nn) * 40];
      bfm[tj] = load_frag(r + half * 8, r + 16 + half * 8);
    }
#pragma unroll
    for (int ti = 0; ti < 2; ++ti)
#pragma unroll
      for (int tj = 0; tj < 4; ++tj)
        acc[ti][tj] = wmma_bf16(af[ti], bfm[tj], acc[ti][tj]);

    __syncthreads();  // done reading buf[cur] before TDM stage s+2 overwrites it
  }

#pragma unroll
  for (int ti = 0; ti < 2; ++ti)
#pragma unroll
    for (int tj = 0; tj < 4; ++tj) {
      const int col = colb + wn * 64 + tj * 16 + nn;
      const float bi = bias[col];
#pragma unroll
      for (int r = 0; r < 8; ++r) {
        const int row = rowb + wm * 32 + ti * 16 + r + half * 8;
        float v = acc[ti][tj][r] + bi;
        if (act == 1) v = 0.5f * v * (1.0f + erff(v * 0.70710678f));
        const size_t idx = (size_t)row * N + col;
        if (outb) outb[idx] = (bf16)v;
        if (outf) outf[idx] = resid ? (v + resid[idx]) : v;
      }
    }
}

// ---------------- attention: per (head, 128-query block) ----------------
// Quirk handled exactly: masked (p>q) scores are 1e-9 and DO enter softmax.
__global__ __launch_bounds__(256) void attn_kernel(const bf16* __restrict__ Q,
                                                   const bf16* __restrict__ Kb,
                                                   const bf16* __restrict__ Vb,
                                                   float* __restrict__ Y) {
  __shared__ __align__(16) bf16 Kc[32 * 64];        // [key][hd]
  __shared__ __align__(16) bf16 Vt[64 * 40];        // [hd][key] padded
  __shared__ __align__(16) bf16 Pbuf[8 * 16 * 40];  // per-wave P (16x32) padded

  const int tid  = threadIdx.x;
  const int lane = tid & 31;
  const int wid  = tid >> 5;
  const int half = lane >> 4;
  const int nn   = lane & 15;

  const int b    = blockIdx.y >> 4;
  const int head = blockIdx.y & 15;
  const int qbase = blockIdx.x * 128 + wid * 16;

  const bf16* qrow = Q + ((size_t)(b * Tc + qbase + nn) * Hc + head * HD);
  v16bf aq[2];
#pragma unroll
  for (int kk = 0; kk < 2; ++kk)
    aq[kk] = load_frag(qrow + kk * 32 + half * 8, qrow + kk * 32 + 16 + half * 8);

  float mrun[8], lrun[8];
  v8f acc[4];
#pragma unroll
  for (int r = 0; r < 8; ++r) { mrun[r] = -1e30f; lrun[r] = 0.f; }
#pragma unroll
  for (int j = 0; j < 4; ++j)
#pragma unroll
    for (int r = 0; r < 8; ++r) acc[j][r] = 0.f;

  bf16* Pw = Pbuf + wid * 16 * 40;

  for (int c = 0; c < Tc / 32; ++c) {
    const int cb = c * 32;
    __syncthreads();
    {  // stage 32 keys of K (row-major) and V (transposed) into LDS
      int key = tid >> 3;
      int h8  = (tid & 7) * 8;
      const size_t base = (size_t)(b * Tc + cb + key) * Hc + head * HD + h8;
      bf16x8 kv = *(const bf16x8*)(Kb + base);
      *(bf16x8*)(Kc + key * 64 + h8) = kv;
      bf16x8 vv = *(const bf16x8*)(Vb + base);
#pragma unroll
      for (int j = 0; j < 8; ++j) Vt[(h8 + j) * 40 + key] = vv[j];
    }
    __syncthreads();

    // scores: two 16x16 tiles covering keys [cb, cb+32)
    v8f s[2];
#pragma unroll
    for (int t2 = 0; t2 < 2; ++t2) {
#pragma unroll
      for (int r = 0; r < 8; ++r) s[t2][r] = 0.f;
#pragma unroll
      for (int kk = 0; kk < 2; ++kk) {
        const bf16* rptr = Kc + (t2 * 16 + nn) * 64 + kk * 32;
        v16bf bk = load_frag(rptr + half * 8, rptr + 16 + half * 8);
        s[t2] = wmma_bf16(aq[kk], bk, s[t2]);
      }
    }

    // scale + mask-quirk + online softmax update
#pragma unroll
    for (int r = 0; r < 8; ++r) {
      const int qg = qbase + r + half * 8;
#pragma unroll
      for (int t2 = 0; t2 < 2; ++t2) {
        const int pg = cb + t2 * 16 + nn;
        float sv = s[t2][r];
        sv = (pg > qg) ? 1e-9f : sv * 0.125f;  // 1/sqrt(64)
        s[t2][r] = sv;
      }
      float cm = fmaxf(s[0][r], s[1][r]);
#pragma unroll
      for (int m = 1; m <= 8; m <<= 1) cm = fmaxf(cm, __shfl_xor(cm, m, 32));
      const float nm = fmaxf(mrun[r], cm);
      const float f  = __expf(mrun[r] - nm);
      mrun[r] = nm;
      const float p0 = __expf(s[0][r] - nm);
      const float p1 = __expf(s[1][r] - nm);
      float rsum = p0 + p1;
#pragma unroll
      for (int m = 1; m <= 8; m <<= 1) rsum += __shfl_xor(rsum, m, 32);
      lrun[r] = lrun[r] * f + rsum;
#pragma unroll
      for (int j = 0; j < 4; ++j) acc[j][r] *= f;
      Pw[(r + half * 8) * 40 + nn]      = (bf16)p0;
      Pw[(r + half * 8) * 40 + 16 + nn] = (bf16)p1;
    }
    asm volatile("s_wait_dscnt 0" ::: "memory");  // per-wave LDS RAW fence

    // P(16x32) @ V(32x64): 4 WMMAs over hd column tiles
    const bf16* prow = Pw + nn * 40;
    v16bf pa = load_frag(prow + half * 8, prow + 16 + half * 8);
#pragma unroll
    for (int j = 0; j < 4; ++j) {
      const bf16* vr = Vt + (j * 16 + nn) * 40;
      v16bf bv = load_frag(vr + half * 8, vr + 16 + half * 8);
      acc[j] = wmma_bf16(pa, bv, acc[j]);
    }
  }

  // normalize and write y (fp32, (B,T,H) layout)
#pragma unroll
  for (int j = 0; j < 4; ++j)
#pragma unroll
    for (int r = 0; r < 8; ++r) {
      const int t  = qbase + r + half * 8;
      const int hd = j * 16 + nn;
      Y[(size_t)(b * Tc + t) * Hc + head * HD + hd] = acc[j][r] / lrun[r];
    }
}

// ---------------- host launcher ----------------
extern "C" void kernel_launch(void* const* d_in, const int* in_sizes, int n_in,
                              void* d_out, int out_size, void* d_ws, size_t ws_size,
                              hipStream_t stream) {
  const float* x     = (const float*)d_in[0];
  const float* ln1_g = (const float*)d_in[1];
  const float* ln1_b = (const float*)d_in[2];
  const float* ln2_g = (const float*)d_in[3];
  const float* ln2_b = (const float*)d_in[4];
  const float* Wq    = (const float*)d_in[5];
  const float* bq    = (const float*)d_in[6];
  const float* Wk    = (const float*)d_in[7];
  const float* bk    = (const float*)d_in[8];
  const float* Wv    = (const float*)d_in[9];
  const float* bv    = (const float*)d_in[10];
  const float* W1    = (const float*)d_in[11];
  const float* b1    = (const float*)d_in[12];
  const float* W2    = (const float*)d_in[13];
  const float* b2    = (const float*)d_in[14];
  float* out = (float*)d_out;

  // workspace layout (all offsets 16B-aligned)
  char* p = (char*)d_ws;
  auto take = [&](size_t bytes) { char* q = p; p += bytes; return q; };
  const size_t HH2   = (size_t)Hc * Hc * 2;
  const size_t HH4_2 = (size_t)Hc * H4 * 2;
  const size_t MH2   = (size_t)Mrows * Hc * 2;
  const size_t MH4_2 = (size_t)Mrows * H4 * 2;
  const size_t MHf   = (size_t)Mrows * Hc * 4;
  bf16*  wq_t = (bf16*)take(HH2);    // Hc x Hc  (transposed)
  bf16*  wk_t = (bf16*)take(HH2);
  bf16*  wv_t = (bf16*)take(HH2);
  bf16*  w1_t = (bf16*)take(HH4_2);  // H4 x Hc
  bf16*  w2_t = (bf16*)take(HH4_2);  // Hc x H4
  bf16*  h1_b = (bf16*)take(MH2);
  bf16*  q_b  = (bf16*)take(MH2);
  bf16*  k_b  = (bf16*)take(MH2);
  bf16*  v_b  = (bf16*)take(MH2);
  bf16*  m1_b = (bf16*)take(MH4_2);
  bf16*  h2_b = (bf16*)take(MH2);
  float* y_f  = (float*)take(MHf);
  float* x1_f = (float*)take(MHf);

  const int nHH = Hc * Hc, nH4 = Hc * H4, nMH = Mrows * Hc;
  // 1. weights -> bf16, transposed (so TDM can stage contiguous-row tiles)
  cvt_t_kernel<<<(nHH + 255) / 256, 256, 0, stream>>>(Wq, wq_t, Hc, Hc);
  cvt_t_kernel<<<(nHH + 255) / 256, 256, 0, stream>>>(Wk, wk_t, Hc, Hc);
  cvt_t_kernel<<<(nHH + 255) / 256, 256, 0, stream>>>(Wv, wv_t, Hc, Hc);
  cvt_t_kernel<<<(nH4 + 255) / 256, 256, 0, stream>>>(W1, w1_t, Hc, H4);
  cvt_t_kernel<<<(nH4 + 255) / 256, 256, 0, stream>>>(W2, w2_t, H4, Hc);
  // 2. LN1
  ln_kernel<<<Mrows, 256, 0, stream>>>(x, ln1_g, ln1_b, h1_b);
  // 3. QKV projections (bf16 out)
  dim3 gq(Hc / 128, Mrows / 128);
  gemm_bf16_kernel<<<gq, 256, 0, stream>>>(h1_b, wq_t, bq, nullptr, q_b, nullptr,
                                           Mrows, Hc, Hc, 0);
  gemm_bf16_kernel<<<gq, 256, 0, stream>>>(h1_b, wk_t, bk, nullptr, k_b, nullptr,
                                           Mrows, Hc, Hc, 0);
  gemm_bf16_kernel<<<gq, 256, 0, stream>>>(h1_b, wv_t, bv, nullptr, v_b, nullptr,
                                           Mrows, Hc, Hc, 0);
  // 4. attention
  dim3 ga(Tc / 128, Bc * Kh);
  attn_kernel<<<ga, 256, 0, stream>>>(q_b, k_b, v_b, y_f);
  // 5. residual 1
  add_kernel<<<(nMH + 255) / 256, 256, 0, stream>>>(x, y_f, x1_f, nMH);
  // 6. LN2
  ln_kernel<<<Mrows, 256, 0, stream>>>(x1_f, ln2_g, ln2_b, h2_b);
  // 7. MLP up + GELU (bf16 out)
  dim3 g1(H4 / 128, Mrows / 128);
  gemm_bf16_kernel<<<g1, 256, 0, stream>>>(h2_b, w1_t, b1, nullptr, m1_b, nullptr,
                                           Mrows, H4, Hc, 1);
  // 8. MLP down + residual 2 -> d_out (fp32)
  dim3 g2(Hc / 128, Mrows / 128);
  gemm_bf16_kernel<<<g2, 256, 0, stream>>>(m1_b, w2_t, b2, x1_f, nullptr, out,
                                           Mrows, Hc, H4, 0);
  (void)in_sizes; (void)n_in; (void)out_size; (void)ws_size;
}
